// GraphNewPolicyNetwork_33827162423529
// MI455X (gfx1250) — matrix operand order repaired and checked by reading, not compiled, not run on hardware
//
#include <hip/hip_runtime.h>
#include <cstdint>
#include <cstddef>

typedef __attribute__((ext_vector_type(16))) _Float16 v16h;
typedef __attribute__((ext_vector_type(8)))  float    v8f;

// ---------------------------------------------------------------------------
// Degree / normalization:  deg = 2 + sum_incoming(1);  dinv = rsqrt(deg)
// ---------------------------------------------------------------------------
__global__ void init_deg_kernel(float* __restrict__ deg, int N) {
  int i = blockIdx.x * blockDim.x + threadIdx.x;
  if (i < N) deg[i] = 2.0f;
}

__global__ void deg_accum_kernel(const int* __restrict__ dst, float* __restrict__ deg, int E) {
  int i = blockIdx.x * blockDim.x + threadIdx.x;
  if (i < E) atomicAdd(&deg[dst[i]], 1.0f);
}

__global__ void dinv_kernel(float* __restrict__ deg, int N) {
  int i = blockIdx.x * blockDim.x + threadIdx.x;
  if (i < N) deg[i] = rsqrtf(deg[i]);
}

// ---------------------------------------------------------------------------
// WMMA GEMM: out[N x NOUT] = A[N x K] @ W[K x NOUT]    (f16 math, f32 accum)
// A is either one dense buffer (lda = K) or, when A1 != nullptr, the virtual
// concat [A0 (N x 64) | A1 (N x 64)]  (K must be 128 in that case).
// W staged to LDS transposed (col-major rows) so each B fragment is 16
// contiguous halfs -> two ds_load_b128 per tile.
// One wave = 16 output rows; 8 waves / 256-thread block = 128 rows.
// ---------------------------------------------------------------------------
template<int K, int NOUT, bool BIASRELU>
__global__ __launch_bounds__(256) void wmma_gemm_kernel(
    const float* __restrict__ A0, const float* __restrict__ A1,
    const float* __restrict__ W,  const float* __restrict__ bias,
    float* __restrict__ out, int N)
{
  __shared__ __align__(32) _Float16 ldsW[NOUT * K];  // Wt[col][k]
  const int tid = threadIdx.x;
  for (int i = tid; i < K * NOUT; i += 256) {
    const int k = i % K, n = i / K;
    ldsW[n * K + k] = (_Float16)W[(size_t)k * NOUT + n];
  }
  __syncthreads();

  const int wid  = tid >> 5;
  const int lane = tid & 31;
  const int m0   = (blockIdx.x * 8 + wid) * 16;
  if (m0 >= N) return;

  const bool lo   = lane < 16;
  const int  mrow = m0 + (lane & 15);
  const int  rowc = mrow < N ? mrow : N - 1;   // clamp tail-tile loads

  constexpr int NT = NOUT / 16;   // output column tiles
  constexpr int NC = K / 32;      // K chunks of 32
  v8f acc[NT];
#pragma unroll
  for (int t = 0; t < NT; ++t) acc[t] = (v8f){};

#pragma unroll
  for (int c = 0; c < NC; ++c) {
    // ---- A fragment (16x32 f16, contiguous k = 16h + s, same as B) ----
    const float* srcp; int ldx, kloc;
    if (A1 != nullptr && c * 32 >= 64) { srcp = A1; ldx = 64; kloc = c * 32 - 64; }
    else                               { srcp = A0; ldx = (A1 != nullptr) ? 64 : K; kloc = c * 32; }
    const float* rp = srcp + (size_t)rowc * ldx + kloc + (lo ? 0 : 16);
    const float4 f0 = *(const float4*)(rp + 0);    // K kb..kb+3
    const float4 f1 = *(const float4*)(rp + 4);    // K kb+4..kb+7
    const float4 f2 = *(const float4*)(rp + 8);    // K kb+8..kb+11
    const float4 f3 = *(const float4*)(rp + 12);   // K kb+12..kb+15
    v16h a;
    a[0]=(_Float16)f0.x; a[1]=(_Float16)f0.y; a[2]=(_Float16)f0.z; a[3]=(_Float16)f0.w;
    a[4]=(_Float16)f1.x; a[5]=(_Float16)f1.y; a[6]=(_Float16)f1.z; a[7]=(_Float16)f1.w;
    a[8]=(_Float16)f2.x; a[9]=(_Float16)f2.y; a[10]=(_Float16)f2.z; a[11]=(_Float16)f2.w;
    a[12]=(_Float16)f3.x; a[13]=(_Float16)f3.y; a[14]=(_Float16)f3.z; a[15]=(_Float16)f3.w;

    const int kstart = c * 32 + (lo ? 0 : 16);     // B: lanes 0-15 K[0..15], 16-31 K[16..31]
#pragma unroll
    for (int t = 0; t < NT; ++t) {
      const int col = t * 16 + (lane & 15);
      const v16h b = *(const v16h*)&ldsW[col * K + kstart];
      acc[t] = __builtin_amdgcn_wmma_f32_16x16x32_f16(
          /*neg_a=*/false, a, /*neg_b=*/false, b,
          /*c_mod=*/(short)0, acc[t], /*reuse_a=*/false, /*reuse_b=*/false);
    }
  }

  // ---- epilogue: VGPR d -> row m0 + d (+8 for upper half-wave) ----
#pragma unroll
  for (int t = 0; t < NT; ++t) {
    const int col = t * 16 + (lane & 15);
#pragma unroll
    for (int d = 0; d < 8; ++d) {
      const int row = m0 + d + (lo ? 0 : 8);
      if (row < N) {
        float v = acc[t][d];
        if constexpr (BIASRELU) { v += bias[col]; v = fmaxf(v, 0.0f); }
        out[(size_t)row * NOUT + col] = v;
      }
    }
  }
}

// ---------------------------------------------------------------------------
// GCN aggregation (H = 64 features)
// ---------------------------------------------------------------------------
__global__ void self_loop_init_kernel(const float* __restrict__ xw,
                                      const float* __restrict__ dinv,
                                      float* __restrict__ agg, int total) {
  int i = blockIdx.x * blockDim.x + threadIdx.x;
  if (i < total) {
    const float di = dinv[i >> 6];
    agg[i] = 2.0f * di * di * xw[i];     // self-loop weight 2 / deg
  }
}

__global__ void agg_edges_kernel(const int* __restrict__ src, const int* __restrict__ dst,
                                 const float* __restrict__ dinv,
                                 const float* __restrict__ xw, float* __restrict__ agg,
                                 long long total) {
  const long long i = (long long)blockIdx.x * blockDim.x + threadIdx.x;
  if (i >= total) return;
  const int e = (int)(i >> 6);
  const int h = (int)(i & 63);
  const int s = src[e], d = dst[e];
  const float norm = dinv[s] * dinv[d];
  atomicAdd(&agg[(size_t)d * 64 + h], xw[(size_t)s * 64 + h] * norm);
}

__global__ void bias_relu_kernel(float* __restrict__ buf, const float* __restrict__ bias, int total) {
  int i = blockIdx.x * blockDim.x + threadIdx.x;
  if (i < total) buf[i] = fmaxf(buf[i] + bias[i & 63], 0.0f);
}

// ---------------------------------------------------------------------------
// Head: prob = sigmoid(z2 @ Wo + bo), one wave32 per node
// ---------------------------------------------------------------------------
__global__ void head_sigmoid_kernel(const float* __restrict__ z2,
                                    const float* __restrict__ Wo,
                                    const float* __restrict__ bo,
                                    float* __restrict__ prob, int N) {
  const int gid  = blockIdx.x * blockDim.x + threadIdx.x;
  const int n    = gid >> 5;
  const int lane = gid & 31;
  if (n >= N) return;
  float s = z2[(size_t)n * 64 + lane]      * Wo[lane]
          + z2[(size_t)n * 64 + 32 + lane] * Wo[32 + lane];
#pragma unroll
  for (int off = 16; off > 0; off >>= 1) s += __shfl_xor(s, off, 32);
  if (lane == 0) prob[n] = 1.0f / (1.0f + expf(-(s + bo[0])));
}

// ---------------------------------------------------------------------------
// JAX threefry2x32 (key = jax.random.key(42) -> {0, 42}), exact key schedule,
// matching jax.random.uniform(key, (N,2), minval=1e-20, maxval=1.0)
// ---------------------------------------------------------------------------
__device__ __forceinline__ void tf_round(uint32_t& x0, uint32_t& x1, int r) {
  x0 += x1; x1 = (x1 << r) | (x1 >> (32 - r)); x1 ^= x0;
}

__device__ __forceinline__ void threefry2x32_42(uint32_t c0, uint32_t c1,
                                                uint32_t& o0, uint32_t& o1) {
  const uint32_t k0 = 0u, k1 = 42u, k2 = k0 ^ k1 ^ 0x1BD11BDAu;
  uint32_t x0 = c0 + k0, x1 = c1 + k1;
  tf_round(x0,x1,13); tf_round(x0,x1,15); tf_round(x0,x1,26); tf_round(x0,x1,6);
  x0 += k1; x1 += k2 + 1u;
  tf_round(x0,x1,17); tf_round(x0,x1,29); tf_round(x0,x1,16); tf_round(x0,x1,24);
  x0 += k2; x1 += k0 + 2u;
  tf_round(x0,x1,13); tf_round(x0,x1,15); tf_round(x0,x1,26); tf_round(x0,x1,6);
  x0 += k0; x1 += k1 + 3u;
  tf_round(x0,x1,17); tf_round(x0,x1,29); tf_round(x0,x1,16); tf_round(x0,x1,24);
  x0 += k1; x1 += k2 + 4u;
  tf_round(x0,x1,13); tf_round(x0,x1,15); tf_round(x0,x1,26); tf_round(x0,x1,6);
  x0 += k2; x1 += k0 + 5u;
  o0 = x0; o1 = x1;
}

__device__ __forceinline__ uint32_t jax_bits(uint32_t f, uint32_t total) {
  const uint32_t half = total >> 1;          // total = 2N is even
  uint32_t o0, o1;
  if (f < half) { threefry2x32_42(f, f + half, o0, o1); return o0; }
  threefry2x32_42(f - half, f, o0, o1); return o1;
}

__device__ __forceinline__ float gumbel_draw(uint32_t bits) {
  const uint32_t m = (bits >> 9) | 0x3F800000u;
  float u; __builtin_memcpy(&u, &m, 4);
  u -= 1.0f;
  u = u * (1.0f - 1e-20f) + 1e-20f;
  u = fmaxf(u, 1e-20f);
  return -logf(-logf(u));
}

__global__ void gumbel_kernel(const float* __restrict__ prob, float* __restrict__ ind, int N) {
  int n = blockIdx.x * blockDim.x + threadIdx.x;
  if (n >= N) return;
  const float p  = prob[n];
  const float g0 = gumbel_draw(jax_bits(2u * (uint32_t)n,      2u * (uint32_t)N));
  const float g1 = gumbel_draw(jax_bits(2u * (uint32_t)n + 1u, 2u * (uint32_t)N));
  // argmax over softmax(logits + g) == argmax over logits + g; argmax ties -> index 0
  ind[n] = ((p + g1) > (1.0f - p + g0)) ? 1.0f : 0.0f;
}

// ---------------------------------------------------------------------------
// Launch
// ---------------------------------------------------------------------------
extern "C" void kernel_launch(void* const* d_in, const int* in_sizes, int n_in,
                              void* d_out, int out_size, void* d_ws, size_t ws_size,
                              hipStream_t stream) {
  const float* rep = (const float*)d_in[0];
  const int*  eidx = (const int*)  d_in[1];
  const float* W1  = (const float*)d_in[2];
  const float* b1  = (const float*)d_in[3];
  const float* W2  = (const float*)d_in[4];
  const float* b2  = (const float*)d_in[5];
  const float* Wa  = (const float*)d_in[6];
  const float* ba  = (const float*)d_in[7];
  const float* Wb  = (const float*)d_in[8];
  const float* bb  = (const float*)d_in[9];
  const float* Wo  = (const float*)d_in[10];
  const float* bo  = (const float*)d_in[11];

  const int N = in_sizes[0] / 64;
  const int E = in_sizes[1] / 2;
  const int* src = eidx;
  const int* dst = eidx + E;

  // workspace: dinv[N] | bufA[N x 128] | bufB[N x 64]   (~77 MB for N=100k)
  float* dinv = (float*)d_ws;
  float* bufA = dinv + N;
  float* bufB = bufA + (size_t)N * 128;

  float* prob = (float*)d_out;       // out[0..N)  = prob
  float* ind  = prob + N;            // out[N..2N) = hard gumbel sample

  const int TB   = 256;
  const int gN   = (N + TB - 1) / TB;
  const int gE   = (E + TB - 1) / TB;
  const int gNH  = (N * 64 + TB - 1) / TB;
  const long long eh = (long long)E * 64;
  const int gEH  = (int)((eh + TB - 1) / TB);
  const int gMM  = (N + 127) / 128;            // 8 waves x 16 rows per block
  const int gHD  = (N * 32 + TB - 1) / TB;     // one wave per node

  // degree -> dinv
  init_deg_kernel<<<gN, TB, 0, stream>>>(dinv, N);
  deg_accum_kernel<<<gE, TB, 0, stream>>>(dst, dinv, E);
  dinv_kernel<<<gN, TB, 0, stream>>>(dinv, N);

  // GCN layer 1
  wmma_gemm_kernel<64, 64, false><<<gMM, TB, 0, stream>>>(rep, nullptr, W1, nullptr, bufA, N);
  self_loop_init_kernel<<<gNH, TB, 0, stream>>>(bufA, dinv, bufB, N * 64);
  agg_edges_kernel<<<gEH, TB, 0, stream>>>(src, dst, dinv, bufA, bufB, eh);
  bias_relu_kernel<<<gNH, TB, 0, stream>>>(bufB, b1, N * 64);           // bufB = h1

  // GCN layer 2
  wmma_gemm_kernel<64, 64, false><<<gMM, TB, 0, stream>>>(bufB, nullptr, W2, nullptr, bufA, N);
  self_loop_init_kernel<<<gNH, TB, 0, stream>>>(bufA, dinv, bufB, N * 64);
  agg_edges_kernel<<<gEH, TB, 0, stream>>>(src, dst, dinv, bufA, bufB, eh);
  bias_relu_kernel<<<gNH, TB, 0, stream>>>(bufB, b2, N * 64);           // bufB = h2

  // MLP head: z1 = relu([rep | h2] @ Wa + ba); z2 = relu(z1 @ Wb + bb)
  wmma_gemm_kernel<128, 128, true><<<gMM, TB, 0, stream>>>(rep, bufB, Wa, ba, bufA, N);
  wmma_gemm_kernel<128, 64,  true><<<gMM, TB, 0, stream>>>(bufA, nullptr, Wb, bb, bufB, N);

  // prob + hard gumbel sample
  head_sigmoid_kernel<<<gHD, TB, 0, stream>>>(bufB, Wo, bo, prob, N);
  gumbel_kernel<<<gN, TB, 0, stream>>>(prob, ind, N);
}
